// Model_39676907885998
// MI455X (gfx1250) — matrix operand order repaired and checked by reading, not compiled, not run on hardware
//
#include <hip/hip_runtime.h>
#include <math.h>
#include <stdint.h>

typedef __attribute__((ext_vector_type(16))) _Float16 v16h;
typedef __attribute__((ext_vector_type(8)))  float    v8f;

constexpr int Bsz = 2;
constexpr int Hn  = 16;
constexpr int Sq  = 2048;
constexpr int Dh  = 128;

constexpr int KB  = 64;          // key block per iteration (fast path)
constexpr int QW  = 16;          // query rows per wave
constexpr int NW  = 8;           // waves per block
constexpr int BQ  = QW * NW;     // 128 query rows per block
constexpr int NKT = KB / 16;     // 4 N-tiles in QK^T
constexpr int NKS = KB / 32;     // 2 contraction slices in P@V
constexpr float LOG2E = 1.44269504088896340736f;

constexpr int KPAD = 8;          // row pad (halfs): stride 272B -> conflict-free b128 reads
constexpr int VPAD = 8;          // row stride 144B
constexpr int PPAD = 8;          // row stride 144B

// ---------------------------------------------------------------------------
// Prepass: K f32 -> f16 (same layout), V f32 -> f16 transposed [bh][d][s].
// grid = Bsz*Hn*(Sq/32) = 2048 blocks, 256 threads.
// ---------------------------------------------------------------------------
__global__ __launch_bounds__(256)
void cvt_kv_kernel(const float* __restrict__ K, const float* __restrict__ V,
                   _Float16* __restrict__ K16, _Float16* __restrict__ Vt16)
{
    const int tid = threadIdx.x;

    // ---- K: elementwise convert, 4096 elements per block ----
    {
        const size_t base = (size_t)blockIdx.x * 4096;
        #pragma unroll
        for (int i = 0; i < 4; ++i) {
            const size_t e = base + (size_t)(tid + i * 256) * 4;
            float4 k4 = *(const float4*)(K + e);
            K16[e + 0] = (_Float16)k4.x;
            K16[e + 1] = (_Float16)k4.y;
            K16[e + 2] = (_Float16)k4.z;
            K16[e + 3] = (_Float16)k4.w;
        }
    }

    // ---- V: 32(s) x 128(d) tile transpose via LDS ----
    __shared__ _Float16 tile[32][Dh + 8];
    const int bh = blockIdx.x >> 6;           // 64 s-tiles per head
    const int s0 = (blockIdx.x & 63) * 32;
    const float* Vh = V + ((size_t)bh * Sq + s0) * Dh;
    #pragma unroll
    for (int i = 0; i < 4; ++i) {
        const int e   = (tid + i * 256) * 4;
        const int key = e >> 7;
        const int d   = e & 127;
        float4 v4 = *(const float4*)(Vh + (size_t)key * Dh + d);
        tile[key][d + 0] = (_Float16)v4.x;
        tile[key][d + 1] = (_Float16)v4.y;
        tile[key][d + 2] = (_Float16)v4.z;
        tile[key][d + 3] = (_Float16)v4.w;
    }
    __syncthreads();
    const int d  = tid >> 1;
    const int sh = (tid & 1) * 16;
    v16h o;
    #pragma unroll
    for (int j = 0; j < 16; ++j) o[j] = tile[sh + j][d];
    *(v16h*)(Vt16 + ((size_t)bh * Dh + d) * Sq + s0 + sh) = o;
}

// ---------------------------------------------------------------------------
// Fast path: flash attention, f16 WMMA, async global->LDS tile copies,
// double buffered, KB = 64 keys per iteration.
// ---------------------------------------------------------------------------
__global__ __launch_bounds__(256, 1)
void fattn_async_kernel(const float* __restrict__ Q, const _Float16* __restrict__ K16,
                        const _Float16* __restrict__ Vt16, const float* __restrict__ invs,
                        float* __restrict__ O)
{
    __shared__ alignas(16) _Float16 Ksh[2][KB][Dh + KPAD];   // [key][d]
    __shared__ alignas(16) _Float16 Vsh[2][Dh][KB + VPAD];   // [d][key] (transposed)
    __shared__ alignas(16) _Float16 Psh[NW][QW][KB + PPAD];  // per-wave P staging

    const int tid  = threadIdx.x;
    const int w    = tid >> 5;
    const int lane = tid & 31;
    const int half = lane >> 4;
    const int l16  = lane & 15;

    const int bh    = blockIdx.x >> 4;
    const int qblk  = blockIdx.x & 15;
    const int qbase = qblk * BQ + w * QW;

    const float scale = LOG2E / invs[bh];

    const float*    __restrict__ Qh = Q    + (size_t)bh * Sq * Dh;
    const _Float16* __restrict__ Kh = K16  + (size_t)bh * Sq * Dh;
    const _Float16* __restrict__ Vh = Vt16 + (size_t)bh * Dh * Sq;
    float*          __restrict__ Oh = O    + (size_t)bh * Sq * Dh;

    // LDS byte offsets (low 32 bits of the flat shared address = LDS offset)
    const uint32_t ldsK0 = (uint32_t)(uintptr_t)&Ksh[0][0][0];
    const uint32_t ldsV0 = (uint32_t)(uintptr_t)&Vsh[0][0][0];

    // Issue one KB x Dh K-tile + Dh x KB V-tile as async b128 copies (8 per wave)
    auto issue_tile = [&](int kb, int buf) {
        #pragma unroll
        for (int i = 0; i < 4; ++i) {               // K: 1024 16B chunks
            const int c   = tid + i * 256;
            const int key = c >> 4;                 // 16 chunks per 128-half row
            const int dh8 = (c & 15) * 8;
            uint32_t goff = (uint32_t)(((uint32_t)(kb + key) * Dh + dh8) * 2);
            uint32_t loff = ldsK0 + ((uint32_t)(buf * KB + key) * (Dh + KPAD) + dh8) * 2;
            asm volatile("global_load_async_to_lds_b128 %0, %1, %2"
                         :: "v"(loff), "v"(goff), "s"(Kh) : "memory");
        }
        #pragma unroll
        for (int i = 0; i < 4; ++i) {               // V^T: 1024 16B chunks
            const int c   = tid + i * 256;
            const int d   = c >> 3;                 // 8 chunks per 64-half row
            const int sh8 = (c & 7) * 8;
            uint32_t goff = (uint32_t)(((uint32_t)d * Sq + kb + sh8) * 2);
            uint32_t loff = ldsV0 + ((uint32_t)(buf * Dh + d) * (KB + VPAD) + sh8) * 2;
            asm volatile("global_load_async_to_lds_b128 %0, %1, %2"
                         :: "v"(loff), "v"(goff), "s"(Vh) : "memory");
        }
    };

    // ---- Q A-fragments: 4 k-slices of 16x32 f16 ----
    v16h aq[4];
    {
        const float* qrow = Qh + (size_t)(qbase + l16) * Dh;
        #pragma unroll
        for (int ds = 0; ds < 4; ++ds) {
            const int d0 = ds * 32 + 16 * half;
            const int d1 = ds * 32 + 16 * half + 8;
            float4 c0 = *(const float4*)(qrow + d0);
            float4 c1 = *(const float4*)(qrow + d0 + 4);
            float4 c2 = *(const float4*)(qrow + d1);
            float4 c3 = *(const float4*)(qrow + d1 + 4);
            aq[ds][0]  = (_Float16)c0.x; aq[ds][1]  = (_Float16)c0.y;
            aq[ds][2]  = (_Float16)c0.z; aq[ds][3]  = (_Float16)c0.w;
            aq[ds][4]  = (_Float16)c1.x; aq[ds][5]  = (_Float16)c1.y;
            aq[ds][6]  = (_Float16)c1.z; aq[ds][7]  = (_Float16)c1.w;
            aq[ds][8]  = (_Float16)c2.x; aq[ds][9]  = (_Float16)c2.y;
            aq[ds][10] = (_Float16)c2.z; aq[ds][11] = (_Float16)c2.w;
            aq[ds][12] = (_Float16)c3.x; aq[ds][13] = (_Float16)c3.y;
            aq[ds][14] = (_Float16)c3.z; aq[ds][15] = (_Float16)c3.w;
        }
    }

    v8f acc[8] = {};
    float mrow[8], lrow[8];
    #pragma unroll
    for (int r = 0; r < 8; ++r) { mrow[r] = -__builtin_inff(); lrow[r] = 0.0f; }

    constexpr int NIT = Sq / KB;    // 32 iterations
    issue_tile(0, 0);

    for (int it = 0; it < NIT; ++it) {
        const int buf = it & 1;
        if (it + 1 < NIT) {
            issue_tile((it + 1) * KB, buf ^ 1);                 // overlap next tile
            asm volatile("s_wait_asynccnt 0x8" ::: "memory");   // current tile done
        } else {
            asm volatile("s_wait_asynccnt 0x0" ::: "memory");
        }
        __syncthreads();

        // ---- S = Q @ K^T, 4 N-tiles of 16 keys ----
        v8f s[NKT] = {};
        #pragma unroll
        for (int ds = 0; ds < 4; ++ds) {
            #pragma unroll
            for (int nt = 0; nt < NKT; ++nt) {
                v16h bk;
                const _Float16* kp = &Ksh[buf][nt * 16 + l16][ds * 32 + 16 * half];
                #pragma unroll
                for (int j = 0; j < 16; ++j) bk[j] = kp[j];
                s[nt] = __builtin_amdgcn_wmma_f32_16x16x32_f16(false, aq[ds], false, bk,
                                                               (short)0, s[nt], false, false);
            }
        }

        // ---- online softmax over 64 keys (base-2) ----
        #pragma unroll
        for (int r = 0; r < 8; ++r) {
            float a[NKT];
            #pragma unroll
            for (int nt = 0; nt < NKT; ++nt) a[nt] = s[nt][r] * scale;
            float mx = fmaxf(fmaxf(a[0], a[1]), fmaxf(a[2], a[3]));
            #pragma unroll
            for (int off = 8; off >= 1; off >>= 1)
                mx = fmaxf(mx, __shfl_xor(mx, off, 32));
            const float mnew = fmaxf(mrow[r], mx);
            float p[NKT], rs = 0.0f;
            #pragma unroll
            for (int nt = 0; nt < NKT; ++nt) { p[nt] = exp2f(a[nt] - mnew); rs += p[nt]; }
            #pragma unroll
            for (int off = 8; off >= 1; off >>= 1)
                rs += __shfl_xor(rs, off, 32);
            const float corr = exp2f(mrow[r] - mnew);
            lrow[r] = lrow[r] * corr + rs;
            mrow[r] = mnew;
            #pragma unroll
            for (int dt = 0; dt < 8; ++dt) acc[dt][r] *= corr;
            #pragma unroll
            for (int nt = 0; nt < NKT; ++nt)
                Psh[w][r + 8 * half][nt * 16 + l16] = (_Float16)p[nt];
        }

        asm volatile("s_wait_dscnt 0x0" ::: "memory");   // same-wave P store->load

        // ---- P A-fragments (two 16x32 slices) ----
        v16h ap[NKS];
        #pragma unroll
        for (int ks = 0; ks < NKS; ++ks) {
            const _Float16* pr = &Psh[w][l16][ks * 32];
            const int k0 = 16 * half, k1 = 16 * half + 8;
            #pragma unroll
            for (int j = 0; j < 8; ++j) { ap[ks][j] = pr[k0 + j]; ap[ks][8 + j] = pr[k1 + j]; }
        }

        // ---- O += P @ V ----
        #pragma unroll
        for (int dt = 0; dt < 8; ++dt) {
            #pragma unroll
            for (int ks = 0; ks < NKS; ++ks) {
                v16h bv;
                const _Float16* vp = &Vsh[buf][dt * 16 + l16][ks * 32 + 16 * half];
                #pragma unroll
                for (int j = 0; j < 16; ++j) bv[j] = vp[j];
                acc[dt] = __builtin_amdgcn_wmma_f32_16x16x32_f16(false, ap[ks], false, bv,
                                                                 (short)0, acc[dt], false, false);
            }
        }

        __syncthreads();   // protect buf before next iteration's async writes
    }

    // ---- epilogue ----
    #pragma unroll
    for (int r = 0; r < 8; ++r) {
        const float inv_l = 1.0f / lrow[r];
        float* op = Oh + (size_t)(qbase + r + 8 * half) * Dh;
        #pragma unroll
        for (int dt = 0; dt < 8; ++dt)
            op[dt * 16 + l16] = acc[dt][r] * inv_l;
    }
}

// ---------------------------------------------------------------------------
// Fallback (ws too small): round-1 kernel, in-kernel conversion, KB=32.
// ---------------------------------------------------------------------------
__global__ __launch_bounds__(256, 1)
void fattn_fallback_kernel(const float* __restrict__ Q, const float* __restrict__ K,
                           const float* __restrict__ V, const float* __restrict__ invs,
                           float* __restrict__ O)
{
    constexpr int KF = 32;
    __shared__ _Float16 Kf[KF][Dh + 8];
    __shared__ _Float16 Vf[Dh][KF + 8];
    __shared__ _Float16 Pf[NW][QW][KF + 8];

    const int tid  = threadIdx.x;
    const int w    = tid >> 5;
    const int lane = tid & 31;
    const int half = lane >> 4;
    const int l16  = lane & 15;

    const int bh    = blockIdx.x >> 4;
    const int qbase = (blockIdx.x & 15) * BQ + w * QW;
    const float scale = LOG2E / invs[bh];

    const size_t head = (size_t)bh * Sq * Dh;
    const float* Qh = Q + head;
    const float* Kh = K + head;
    const float* Vh = V + head;
    float*       Oh = O + head;

    v16h aq[4];
    {
        const float* qrow = Qh + (size_t)(qbase + l16) * Dh;
        #pragma unroll
        for (int ds = 0; ds < 4; ++ds) {
            const int d0 = ds * 32 + 16 * half;
            const int d1 = ds * 32 + 16 * half + 8;
            #pragma unroll
            for (int j = 0; j < 8; ++j) {
                aq[ds][j]     = (_Float16)qrow[d0 + j];
                aq[ds][8 + j] = (_Float16)qrow[d1 + j];
            }
        }
    }

    v8f acc[8] = {};
    float mrow[8], lrow[8];
    #pragma unroll
    for (int r = 0; r < 8; ++r) { mrow[r] = -__builtin_inff(); lrow[r] = 0.0f; }

    for (int kb = 0; kb < Sq; kb += KF) {
        __syncthreads();
        #pragma unroll
        for (int i = 0; i < 4; ++i) {
            const int e   = (tid + i * 256) * 4;
            const int key = e >> 7;
            const int d   = e & 127;
            float4 k4 = *(const float4*)(Kh + (size_t)(kb + key) * Dh + d);
            Kf[key][d + 0] = (_Float16)k4.x; Kf[key][d + 1] = (_Float16)k4.y;
            Kf[key][d + 2] = (_Float16)k4.z; Kf[key][d + 3] = (_Float16)k4.w;
            float4 v4 = *(const float4*)(Vh + (size_t)(kb + key) * Dh + d);
            Vf[d + 0][key] = (_Float16)v4.x; Vf[d + 1][key] = (_Float16)v4.y;
            Vf[d + 2][key] = (_Float16)v4.z; Vf[d + 3][key] = (_Float16)v4.w;
        }
        __syncthreads();

        v8f s0 = {}, s1 = {};
        #pragma unroll
        for (int ds = 0; ds < 4; ++ds) {
            v16h bk0, bk1;
            const _Float16* kp0 = &Kf[l16][ds * 32 + 16 * half];
            const _Float16* kp1 = &Kf[l16 + 16][ds * 32 + 16 * half];
            #pragma unroll
            for (int j = 0; j < 16; ++j) { bk0[j] = kp0[j]; bk1[j] = kp1[j]; }
            s0 = __builtin_amdgcn_wmma_f32_16x16x32_f16(false, aq[ds], false, bk0, (short)0, s0, false, false);
            s1 = __builtin_amdgcn_wmma_f32_16x16x32_f16(false, aq[ds], false, bk1, (short)0, s1, false, false);
        }

        #pragma unroll
        for (int r = 0; r < 8; ++r) {
            float a = s0[r] * scale, b = s1[r] * scale;
            float mx = fmaxf(a, b);
            #pragma unroll
            for (int off = 8; off >= 1; off >>= 1) mx = fmaxf(mx, __shfl_xor(mx, off, 32));
            const float mnew = fmaxf(mrow[r], mx);
            const float p0 = exp2f(a - mnew), p1 = exp2f(b - mnew);
            const float corr = exp2f(mrow[r] - mnew);
            float rs = p0 + p1;
            #pragma unroll
            for (int off = 8; off >= 1; off >>= 1) rs += __shfl_xor(rs, off, 32);
            lrow[r] = lrow[r] * corr + rs;
            mrow[r] = mnew;
            #pragma unroll
            for (int dt = 0; dt < 8; ++dt) acc[dt][r] *= corr;
            Pf[w][r + 8 * half][l16]      = (_Float16)p0;
            Pf[w][r + 8 * half][16 + l16] = (_Float16)p1;
        }

        asm volatile("s_wait_dscnt 0x0" ::: "memory");

        v16h ap;
        {
            const _Float16* pr = &Pf[w][l16][0];
            const int k0 = 16 * half, k1 = 16 * half + 8;
            #pragma unroll
            for (int j = 0; j < 8; ++j) { ap[j] = pr[k0 + j]; ap[8 + j] = pr[k1 + j]; }
        }
        #pragma unroll
        for (int dt = 0; dt < 8; ++dt) {
            v16h bv;
            const _Float16* vp = &Vf[dt * 16 + l16][16 * half];
            #pragma unroll
            for (int j = 0; j < 16; ++j) bv[j] = vp[j];
            acc[dt] = __builtin_amdgcn_wmma_f32_16x16x32_f16(false, ap, false, bv, (short)0, acc[dt], false, false);
        }
    }

    #pragma unroll
    for (int r = 0; r < 8; ++r) {
        const float inv_l = 1.0f / lrow[r];
        float* op = Oh + (size_t)(qbase + r + 8 * half) * Dh;
        #pragma unroll
        for (int dt = 0; dt < 8; ++dt)
            op[dt * 16 + l16] = acc[dt][r] * inv_l;
    }
}

extern "C" void kernel_launch(void* const* d_in, const int* in_sizes, int n_in,
                              void* d_out, int out_size, void* d_ws, size_t ws_size,
                              hipStream_t stream) {
    const float* Q    = (const float*)d_in[0];
    const float* K    = (const float*)d_in[1];
    const float* V    = (const float*)d_in[2];
    const float* invs = (const float*)d_in[3];
    float* O = (float*)d_out;

    const size_t headElems = (size_t)Bsz * Hn * Sq * Dh;        // 8M
    const size_t need      = headElems * 2 * sizeof(_Float16);  // K16 + Vt16 = 32 MB

    if (d_ws != nullptr && ws_size >= need) {
        _Float16* K16  = (_Float16*)d_ws;
        _Float16* Vt16 = K16 + headElems;
        cvt_kv_kernel<<<Bsz * Hn * (Sq / 32), 256, 0, stream>>>(K, V, K16, Vt16);
        fattn_async_kernel<<<Bsz * Hn * (Sq / BQ), 256, 0, stream>>>(Q, K16, Vt16, invs, O);
    } else {
        fattn_fallback_kernel<<<Bsz * Hn * (Sq / BQ), 256, 0, stream>>>(Q, K, V, invs, O);
    }
}
